// TransformerBlock_33492154974646
// MI455X (gfx1250) — matrix operand order repaired and checked by reading, not compiled, not run on hardware
//
#include <hip/hip_runtime.h>
#include <hip/hip_bf16.h>

typedef __attribute__((ext_vector_type(16))) __bf16 bf16x16;
typedef __attribute__((ext_vector_type(8)))  float  floatx8;

// ---------------------------------------------------------------------------
// WMMA helper: D = A(16x32 bf16) x B(32x16 bf16) + C(16x16 f32)
// ---------------------------------------------------------------------------
__device__ __forceinline__ floatx8 wmma_bf16_f32(bf16x16 a, bf16x16 b, floatx8 c) {
  return __builtin_amdgcn_wmma_f32_16x16x32_bf16(
      /*neg_a=*/false, a, /*neg_b=*/false, b,
      /*c_mod=*/(short)0, c, /*reuse_a=*/false, /*reuse_b=*/false);
}

// A-fragment (16x32, bf16) from a row-major fp32 row pointer.
// ISA layout: lane holds row m=lane%16; half=lane/16 selects K runs
// {half*8..half*8+7} and {16+half*8..16+half*8+7} within the 32-K chunk.
__device__ __forceinline__ bf16x16 load_a_f32(const float* rowbase, int kc, int half) {
  bf16x16 r;
#pragma unroll
  for (int g = 0; g < 2; ++g) {
#pragma unroll
    for (int t = 0; t < 8; ++t) {
      r[g * 8 + t] = (__bf16)rowbase[kc * 32 + g * 16 + half * 8 + t];
    }
  }
  return r;
}

// B-fragment (32x16, bf16) from LDS holding W^T ([n][k], row length ldk).
// ISA layout: lane col n, half=lane/16 holds K runs {half*8..half*8+7} and
// {16+half*8..16+half*8+7}; transposed storage makes each run contiguous.
__device__ __forceinline__ bf16x16 load_b_lds(const __bf16* Wt, int ldk,
                                              int ntile, int kc, int lane) {
  const int half = lane >> 4;
  const int n    = (lane & 15) + ntile * 16;
  const __bf16* p = Wt + n * ldk + kc * 32 + half * 8;
  bf16x16 r;
#pragma unroll
  for (int e = 0; e < 16; ++e) r[e] = p[(e >> 3) * 16 + (e & 7)];
  return r;
}

// Order-preserving fp32 <-> uint32 encoding for atomic max.
__device__ __forceinline__ unsigned encf(float x) {
  unsigned u = __float_as_uint(x);
  return (u & 0x80000000u) ? ~u : (u | 0x80000000u);
}
__device__ __forceinline__ float decf(unsigned u) {
  unsigned b = (u & 0x80000000u) ? (u & 0x7fffffffu) : ~u;
  return __uint_as_float(b);
}

// ---------------------------------------------------------------------------
// init: amax = enc(-inf), den = 0, agg = 0
// ---------------------------------------------------------------------------
__global__ void init_kernel(unsigned* __restrict__ amaxU, float* __restrict__ den,
                            float* __restrict__ agg, size_t NC) {
  size_t i = (size_t)blockIdx.x * blockDim.x + threadIdx.x;
  if (i >= NC) return;
  amaxU[i] = encf(-__builtin_inff());
  den[i]   = 0.f;
  agg[i]   = 0.f;
}

// ---------------------------------------------------------------------------
// Generic [M,128] @ [128,128] GEMM, optional bias / relu / residual skip.
// One wave computes a 16x128 output tile (8 ntiles x 4 K-steps of WMMA).
// ---------------------------------------------------------------------------
__global__ void __launch_bounds__(256)
gemm128_kernel(const float* __restrict__ X, const float* __restrict__ W,
               const float* __restrict__ bias, const float* __restrict__ skip,
               float* __restrict__ Y, int M, int relu_flag) {
  __shared__ __bf16 sWt[128 * 128];  // W^T, bf16: 32 KB
  const int tid = threadIdx.x;
  for (int i = tid; i < 128 * 128; i += blockDim.x) {
    int k = i >> 7, n = i & 127;          // W[k][n] row-major (K,N)
    sWt[n * 128 + k] = (__bf16)W[i];
  }
  __syncthreads();

  const int lane = tid & 31, wave = tid >> 5;
  const int mtiles = (M + 15) >> 4;
  int tile = blockIdx.x * (blockDim.x >> 5) + wave;
  if (tile >= mtiles) tile = mtiles - 1;  // pure recompute: deterministic-safe
  const int row0 = tile * 16;
  const int m = lane & 15, half = lane >> 4, ncol = lane & 15;
  int rm = row0 + m; if (rm >= M) rm = M - 1;
  const float* rowb = X + (size_t)rm * 128;

  bf16x16 af[4];
#pragma unroll
  for (int kc = 0; kc < 4; ++kc) af[kc] = load_a_f32(rowb, kc, half);

#pragma unroll
  for (int nt = 0; nt < 8; ++nt) {
    floatx8 c = {0.f, 0.f, 0.f, 0.f, 0.f, 0.f, 0.f, 0.f};
#pragma unroll
    for (int kc = 0; kc < 4; ++kc)
      c = wmma_bf16_f32(af[kc], load_b_lds(sWt, 128, nt, kc, lane), c);
    const int col = nt * 16 + ncol;
    const float b = bias ? bias[col] : 0.f;
#pragma unroll
    for (int rr = 0; rr < 8; ++rr) {
      int row = row0 + rr + half * 8; if (row >= M) row = M - 1;
      float v = c[rr] + b;
      if (relu_flag) v = fmaxf(v, 0.f);
      if (skip) v += skip[(size_t)row * 128 + col];
      Y[(size_t)row * 128 + col] = v;
    }
  }
}

// ---------------------------------------------------------------------------
// Fused per-edge MLP chain: pos_nn (layer1 VALU, layer2 WMMA) -> delta,
// alphaIn = a_dst[i]-a_src[j]+delta, attn_nn (2x WMMA) -> alpha.
// Emits delta & alpha to HBM; atomicMax(amax). One wave = 16 edges.
// ---------------------------------------------------------------------------
__global__ void __launch_bounds__(64)
edge_alpha_kernel(const float* __restrict__ pos,
                  const int* __restrict__ srcIdx, const int* __restrict__ dstIdx,
                  const float* __restrict__ a_src, const float* __restrict__ a_dst,
                  const float* __restrict__ Wp1, const float* __restrict__ bp1,
                  const float* __restrict__ Wp2, const float* __restrict__ bp2,
                  const float* __restrict__ Wa1, const float* __restrict__ ba1,
                  const float* __restrict__ Wa2, const float* __restrict__ ba2,
                  float* __restrict__ wsDelta, float* __restrict__ wsAlpha,
                  unsigned* __restrict__ amaxU, int E, int Et) {
  __shared__ __bf16 sWp2t[128 * 64];   // Wp2^T (64x128 -> [n=128][k=64])  16 KB
  __shared__ __bf16 sWa1t[64 * 128];   // Wa1^T (128x64 -> [n=64][k=128])  16 KB
  __shared__ __bf16 sWa2t[128 * 64];   // Wa2^T (64x128 -> [n=128][k=64])  16 KB
  __shared__ float  sWp1[3 * 64];
  __shared__ float  sBp1[64], sBp2[128], sBa1[64], sBa2[128];
  __shared__ __bf16 sTile[2][16 * 128];  // per-wave C->A shuffle tile, 2x4 KB

  const int tid = threadIdx.x;
  for (int i = tid; i < 64 * 128; i += blockDim.x) {
    { int k = i >> 7, n = i & 127; sWp2t[n * 64 + k]  = (__bf16)Wp2[i]; }
    { int k = i >> 6, n = i & 63;  sWa1t[n * 128 + k] = (__bf16)Wa1[i]; }
    { int k = i >> 7, n = i & 127; sWa2t[n * 64 + k]  = (__bf16)Wa2[i]; }
  }
  for (int i = tid; i < 3 * 64; i += blockDim.x) sWp1[i] = Wp1[i];
  for (int i = tid; i < 64;  i += blockDim.x) { sBp1[i] = bp1[i]; sBa1[i] = ba1[i]; }
  for (int i = tid; i < 128; i += blockDim.x) { sBp2[i] = bp2[i]; sBa2[i] = ba2[i]; }
  __syncthreads();

  const int lane = tid & 31, wave = tid >> 5;
  const int etiles = (Et + 15) >> 4;
  int tile = blockIdx.x * 2 + wave;
  if (tile >= etiles) tile = etiles - 1;  // repeats last tile: max is idempotent
  const int e0 = tile * 16;
  const int m = lane & 15, half = lane >> 4, ncol = lane & 15;

  int em = e0 + m; if (em >= Et) em = Et - 1;
  int jm, im;
  if (em < E) { jm = srcIdx[em]; im = dstIdx[em]; } else { jm = im = em - E; }

  const float rx = pos[(size_t)im * 3 + 0] - pos[(size_t)jm * 3 + 0];
  const float ry = pos[(size_t)im * 3 + 1] - pos[(size_t)jm * 3 + 1];
  const float rz = pos[(size_t)im * 3 + 2] - pos[(size_t)jm * 3 + 2];

  // pos_nn layer 1 (K=3: VALU) directly in A-fragment layout
  bf16x16 t1f[2];
#pragma unroll
  for (int kc = 0; kc < 2; ++kc) {
#pragma unroll
    for (int g = 0; g < 2; ++g) {
#pragma unroll
      for (int t = 0; t < 8; ++t) {
        const int K = kc * 32 + g * 16 + half * 8 + t;  // 0..63
        float v = rx * sWp1[K] + ry * sWp1[64 + K] + rz * sWp1[128 + K] + sBp1[K];
        t1f[kc][g * 8 + t] = (__bf16)fmaxf(v, 0.f);
      }
    }
  }

  __bf16* dT = &sTile[wave][0];

  // delta = t1 @ Wp2 + bp2  (8 ntiles x 2 K-steps)
  floatx8 dacc[8];
#pragma unroll
  for (int nt = 0; nt < 8; ++nt) {
    floatx8 c = {0.f, 0.f, 0.f, 0.f, 0.f, 0.f, 0.f, 0.f};
#pragma unroll
    for (int kc = 0; kc < 2; ++kc)
      c = wmma_bf16_f32(t1f[kc], load_b_lds(sWp2t, 64, nt, kc, lane), c);
    const float b = sBp2[nt * 16 + ncol];
#pragma unroll
    for (int rr = 0; rr < 8; ++rr) c[rr] += b;
    dacc[nt] = c;
  }
#pragma unroll
  for (int nt = 0; nt < 8; ++nt) {
    const int col = nt * 16 + ncol;
#pragma unroll
    for (int rr = 0; rr < 8; ++rr) {
      const int row = rr + half * 8;
      int er = e0 + row; if (er >= Et) er = Et - 1;
      const float dv = dacc[nt][rr];
      wsDelta[(size_t)er * 128 + col] = dv;
      dT[row * 128 + col] = (__bf16)dv;
    }
  }
  __syncthreads();

  // alphaIn = a_dst[i] - a_src[j] + delta, gathered rows are L2-resident
  const float* adRow = a_dst + (size_t)im * 128;
  const float* asRow = a_src + (size_t)jm * 128;
  bf16x16 aif[4];
#pragma unroll
  for (int kc = 0; kc < 4; ++kc) {
#pragma unroll
    for (int g = 0; g < 2; ++g) {
#pragma unroll
      for (int t = 0; t < 8; ++t) {
        const int K = kc * 32 + g * 16 + half * 8 + t;
        float v = adRow[K] - asRow[K] + (float)dT[m * 128 + K];
        aif[kc][g * 8 + t] = (__bf16)v;
      }
    }
  }
  __syncthreads();

  // t2 = relu(alphaIn @ Wa1 + ba1)  (4 ntiles x 4 K-steps), stash [16][64] in LDS
#pragma unroll
  for (int nt = 0; nt < 4; ++nt) {
    floatx8 c = {0.f, 0.f, 0.f, 0.f, 0.f, 0.f, 0.f, 0.f};
#pragma unroll
    for (int kc = 0; kc < 4; ++kc)
      c = wmma_bf16_f32(aif[kc], load_b_lds(sWa1t, 128, nt, kc, lane), c);
    const int col = nt * 16 + ncol;
    const float b = sBa1[col];
#pragma unroll
    for (int rr = 0; rr < 8; ++rr)
      dT[(rr + half * 8) * 64 + col] = (__bf16)fmaxf(c[rr] + b, 0.f);
  }
  __syncthreads();

  bf16x16 t2f[2];
#pragma unroll
  for (int kc = 0; kc < 2; ++kc) {
#pragma unroll
    for (int g = 0; g < 2; ++g) {
#pragma unroll
      for (int t = 0; t < 8; ++t) {
        const int K = kc * 32 + g * 16 + half * 8 + t;
        t2f[kc][g * 8 + t] = dT[m * 64 + K];
      }
    }
  }

  int di8[8];
#pragma unroll
  for (int rr = 0; rr < 8; ++rr) {
    int er = e0 + rr + half * 8; if (er >= Et) er = Et - 1;
    di8[rr] = (er < E) ? dstIdx[er] : er - E;
  }

  // alpha = t2 @ Wa2 + ba2  (8 ntiles x 2 K-steps) -> store + segment max
#pragma unroll
  for (int nt = 0; nt < 8; ++nt) {
    floatx8 c = {0.f, 0.f, 0.f, 0.f, 0.f, 0.f, 0.f, 0.f};
#pragma unroll
    for (int kc = 0; kc < 2; ++kc)
      c = wmma_bf16_f32(t2f[kc], load_b_lds(sWa2t, 64, nt, kc, lane), c);
    const int col = nt * 16 + ncol;
    const float b = sBa2[col];
#pragma unroll
    for (int rr = 0; rr < 8; ++rr) {
      int er = e0 + rr + half * 8; if (er >= Et) er = Et - 1;
      const float av = c[rr] + b;
      wsAlpha[(size_t)er * 128 + col] = av;
      atomicMax(&amaxU[(size_t)di8[rr] * 128 + col], encf(av));
    }
  }
}

// ---------------------------------------------------------------------------
// ex = exp(alpha - amax[dst]) in place; den[dst] += ex
// ---------------------------------------------------------------------------
__global__ void edge_exp_kernel(float* __restrict__ alpha,
                                const unsigned* __restrict__ amaxU,
                                const int* __restrict__ dstIdx,
                                float* __restrict__ den, size_t total, int E) {
  size_t idx = (size_t)blockIdx.x * blockDim.x + threadIdx.x;
  if (idx >= total) return;
  const int e = (int)(idx >> 7), c = (int)(idx & 127);
  const int di = (e < E) ? dstIdx[e] : e - E;
  const float am = decf(amaxU[(size_t)di * 128 + c]);
  const float x = __expf(alpha[idx] - am);
  alpha[idx] = x;
  atomicAdd(&den[(size_t)di * 128 + c], x);
}

// ---------------------------------------------------------------------------
// agg[dst] += (ex/(den[dst]+1e-16)) * (v[src] + delta)
// ---------------------------------------------------------------------------
__global__ void edge_agg_kernel(const float* __restrict__ ex,
                                const float* __restrict__ delta,
                                const float* __restrict__ v,
                                const float* __restrict__ den,
                                const int* __restrict__ srcIdx,
                                const int* __restrict__ dstIdx,
                                float* __restrict__ agg, size_t total, int E) {
  size_t idx = (size_t)blockIdx.x * blockDim.x + threadIdx.x;
  if (idx >= total) return;
  const int e = (int)(idx >> 7), c = (int)(idx & 127);
  int si, di;
  if (e < E) { si = srcIdx[e]; di = dstIdx[e]; } else { si = di = e - E; }
  const float w = ex[idx] / (den[(size_t)di * 128 + c] + 1e-16f);
  const float contrib = w * (v[(size_t)si * 128 + c] + delta[idx]);
  atomicAdd(&agg[(size_t)di * 128 + c], contrib);
}

// ---------------------------------------------------------------------------
// Host driver
// ---------------------------------------------------------------------------
extern "C" void kernel_launch(void* const* d_in, const int* in_sizes, int n_in,
                              void* d_out, int out_size, void* d_ws, size_t ws_size,
                              hipStream_t stream) {
  const float* x     = (const float*)d_in[0];
  const float* pos   = (const float*)d_in[1];
  const int*   ei    = (const int*)d_in[2];
  const float* W_in  = (const float*)d_in[3];
  const float* b_in  = (const float*)d_in[4];
  const float* W_out = (const float*)d_in[5];
  const float* b_out = (const float*)d_in[6];
  const float* W_lin = (const float*)d_in[7];
  const float* W_src = (const float*)d_in[8];
  const float* W_dst = (const float*)d_in[9];
  const float* Wp1   = (const float*)d_in[10];
  const float* bp1   = (const float*)d_in[11];
  const float* Wp2   = (const float*)d_in[12];
  const float* bp2   = (const float*)d_in[13];
  const float* Wa1   = (const float*)d_in[14];
  const float* ba1   = (const float*)d_in[15];
  const float* Wa2   = (const float*)d_in[16];
  const float* ba2   = (const float*)d_in[17];

  const int N  = in_sizes[0] / 128;
  const int E  = in_sizes[2] / 2;
  const int Et = E + N;
  const int* srcIdx = ei;          // edge_index[0]
  const int* dstIdx = ei + E;      // edge_index[1]

  const size_t NC  = (size_t)N * 128;
  const size_t EtC = (size_t)Et * 128;
  float* p = (float*)d_ws;
  float* h      = p; p += NC;
  float* a_src  = p; p += NC;
  float* a_dst  = p; p += NC;
  float* v      = p; p += NC;
  float* den    = p; p += NC;
  float* agg    = p; p += NC;
  unsigned* amaxU = (unsigned*)p; p += NC;
  float* delta  = p; p += EtC;
  float* alpha  = p; p += EtC;
  float* out = (float*)d_out;
  (void)n_in; (void)out_size; (void)ws_size;

  { // softmax-state init (re-zeroed every call: graph-replay safe)
    int thr = 256; int blk = (int)((NC + thr - 1) / thr);
    init_kernel<<<blk, thr, 0, stream>>>(amaxU, den, agg, NC);
  }

  const int mtiles  = (N + 15) >> 4;
  const int gblocks = (mtiles + 7) / 8;  // 8 waves / block
  gemm128_kernel<<<gblocks, 256, 0, stream>>>(x, W_in,  b_in,   nullptr, h,     N, 1);
  gemm128_kernel<<<gblocks, 256, 0, stream>>>(h, W_src, nullptr, nullptr, a_src, N, 0);
  gemm128_kernel<<<gblocks, 256, 0, stream>>>(h, W_dst, nullptr, nullptr, a_dst, N, 0);
  gemm128_kernel<<<gblocks, 256, 0, stream>>>(h, W_lin, nullptr, nullptr, v,     N, 0);

  const int etiles  = (Et + 15) >> 4;
  const int eblocks = (etiles + 1) / 2;  // 2 waves / block (LDS-bound)
  edge_alpha_kernel<<<eblocks, 64, 0, stream>>>(pos, srcIdx, dstIdx, a_src, a_dst,
      Wp1, bp1, Wp2, bp2, Wa1, ba1, Wa2, ba2, delta, alpha, amaxU, E, Et);

  {
    int thr = 256; int blk = (int)((EtC + thr - 1) / thr);
    edge_exp_kernel<<<blk, thr, 0, stream>>>(alpha, amaxU, dstIdx, den, EtC, E);
    edge_agg_kernel<<<blk, thr, 0, stream>>>(alpha, delta, v, den, srcIdx, dstIdx,
                                             agg, EtC, E);
  }

  gemm128_kernel<<<gblocks, 256, 0, stream>>>(agg, W_out, b_out, x, out, N, 1);
}
